// GATNet_27195732918377
// MI455X (gfx1250) — compile-verified, matrix operand order checked by reading, not run on hardware
//
#include <hip/hip_runtime.h>
#include <hip/hip_bf16.h>

#define N_NODES 102600
#define T_STEPS 64
#define F_IN    5
#define HDIM    64
#define N_EDGES (16 * N_NODES)
#define E_TOT   (N_EDGES + N_NODES)

typedef __attribute__((ext_vector_type(16))) __bf16 v16bf;
typedef __attribute__((ext_vector_type(8)))  float  v8f;

union BFrag { v16bf v; uint4 u[2]; };

__device__ __forceinline__ unsigned short f2bf(float f) {
    union { float f; unsigned u; } x; x.f = f;
    unsigned r = x.u + 0x7fffu + ((x.u >> 16) & 1u);   // round-to-nearest-even
    return (unsigned short)(r >> 16);
}
__device__ __forceinline__ float bf2f(unsigned short h) {
    union { unsigned u; float f; } x; x.u = ((unsigned)h) << 16;
    return x.f;
}
__device__ __forceinline__ float sigm(float x) {
    return __builtin_amdgcn_rcpf(1.f + __expf(-x));
}
__device__ __forceinline__ float tanh_fast(float x) {
    return 1.f - 2.f * __builtin_amdgcn_rcpf(1.f + __expf(2.f * x));
}

// ---------------------------------------------------------------------------
// Kernel 1: fused LSTM over T=64 steps. 16 nodes / workgroup, 4 waves.
// Wave w owns hidden columns [16w,16w+16): it computes the i/f/g/o WMMA tiles
// for those columns (tiles w, w+4, w+8, w+12) and keeps c-state in registers.
// Augmented GEMM: A=[h|x_t|1|0pad] (K=96), B held in VGPR fragments for all t.
// ---------------------------------------------------------------------------
__global__ __launch_bounds__(128) void lstm_kernel(
    const float* __restrict__ x,   const float* __restrict__ W_ih,
    const float* __restrict__ W_hh,const float* __restrict__ b_ih,
    const float* __restrict__ b_hh,float* __restrict__ hout)
{
    __shared__ alignas(16) unsigned short sB[256][96];           // 48 KB: B col-major
    __shared__ alignas(16) unsigned short sA[16][96];            // 3 KB : augmented A rows
    __shared__ alignas(16) unsigned short sX[16][T_STEPS * F_IN];// 10 KB: per-node x

    const int tid  = threadIdx.x;
    const int lane = tid & 31;
    const int w    = tid >> 5;
    const int nodeBase = blockIdx.x * 16;

    // Build augmented B: rows k=0..63 -> W_hh[col][k]; 64..68 -> W_ih; 69 -> bias; rest 0
    for (int i = tid; i < 256 * 96; i += 128) {
        int col = i / 96, k = i - col * 96;
        float v;
        if      (k < 64)  v = W_hh[col * 64 + k];
        else if (k < 69)  v = W_ih[col * 5 + (k - 64)];
        else if (k == 69) v = b_ih[col] + b_hh[col];
        else              v = 0.f;
        sB[col][k] = f2bf(v);
    }
    // Stage all timesteps of x for this node tile (clamped tail)
    for (int i = tid; i < 16 * T_STEPS * F_IN; i += 128) {
        int node = i / (T_STEPS * F_IN), off = i - node * (T_STEPS * F_IN);
        int gn = nodeBase + node; if (gn > N_NODES - 1) gn = N_NODES - 1;
        sX[node][off] = f2bf(x[(size_t)gn * (T_STEPS * F_IN) + off]);
    }
    // Init A rows: h=0, const-1 at k=69, zero pad
    for (int i = tid; i < 16 * 96; i += 128) {
        int node = i / 96, k = i - node * 96;
        sA[node][k] = (k == 69) ? f2bf(1.0f) : (unsigned short)0;
    }
    __syncthreads();

    const int halfsel = (lane & 16) ? 8 : 0;   // K sub-offset per ISA A/B bf16 layout
    const int lcol    = lane & 15;

    // Preload B fragments into registers: 4 gates x 3 K-chunks (96 VGPRs), reused all 64 steps
    BFrag bfrag[4][3];
#pragma unroll
    for (int g = 0; g < 4; ++g) {
        const int col = 16 * (4 * g + w) + lcol;    // this lane's gate-unit column
#pragma unroll
        for (int c = 0; c < 3; ++c) {
            const int klo = 32 * c + halfsel;
            bfrag[g][c].u[0] = *(const uint4*)&sB[col][klo];
            bfrag[g][c].u[1] = *(const uint4*)&sB[col][klo + 16];
        }
    }

    float cstate[8];
#pragma unroll
    for (int r = 0; r < 8; ++r) cstate[r] = 0.f;

    const int hcol = 16 * w + lcol;             // hidden column this lane updates
    const int mofs = (lane & 16) ? 8 : 0;       // D-tile row half per ISA C/D layout

    for (int t = 0; t < T_STEPS; ++t) {
        // drop x_t into the augmented A rows (cols 64..68; disjoint from h cols)
        if (tid < 16 * F_IN) {
            int node = tid / F_IN, f = tid - node * F_IN;
            sA[node][64 + f] = sX[node][t * F_IN + f];
        }
        __syncthreads();                        // x_t + h_{t-1} visible

        BFrag afrag[3];
#pragma unroll
        for (int c = 0; c < 3; ++c) {
            const int klo = 32 * c + halfsel;
            afrag[c].u[0] = *(const uint4*)&sA[lcol][klo];
            afrag[c].u[1] = *(const uint4*)&sA[lcol][klo + 16];
        }
        __syncthreads();                        // all A-frag reads done before h_t writes

        v8f acc[4];
#pragma unroll
        for (int g = 0; g < 4; ++g) {
            v8f a = {0.f,0.f,0.f,0.f,0.f,0.f,0.f,0.f};
#pragma unroll
            for (int c = 0; c < 3; ++c)
                a = __builtin_amdgcn_wmma_f32_16x16x32_bf16(
                        false, afrag[c].v, false, bfrag[g][c].v,
                        (short)0, a, false, false);
            acc[g] = a;
        }
        // torch gate order i,f,g,o; state update in registers, h_t -> LDS
#pragma unroll
        for (int r = 0; r < 8; ++r) {
            float gi = acc[0][r], gf = acc[1][r], gg = acc[2][r], go = acc[3][r];
            float cn = sigm(gf) * cstate[r] + sigm(gi) * tanh_fast(gg);
            cstate[r] = cn;
            float hn = sigm(go) * tanh_fast(cn);
            sA[mofs + r][hcol] = f2bf(hn);
        }
    }
    __syncthreads();

    for (int i = tid; i < 16 * HDIM; i += 128) {
        int node = i >> 6, col = i & 63;
        int gn = nodeBase + node;
        if (gn < N_NODES) hout[(size_t)gn * HDIM + col] = bf2f(sA[node][col]);
    }
}

// ---------------------------------------------------------------------------
// Kernel 2: xp = h @ gat_w^T (WMMA), plus per-node a_src/a_dst dot products.
// ---------------------------------------------------------------------------
__global__ __launch_bounds__(32) void gat_xp_kernel(
    const float* __restrict__ hbuf, const float* __restrict__ gat_w,
    const float* __restrict__ att_src, const float* __restrict__ att_dst,
    float* __restrict__ xp, float* __restrict__ a_src, float* __restrict__ a_dst)
{
    __shared__ alignas(16) unsigned short sGW[64][64];  // 8 KB (row c = output unit)
    __shared__ alignas(16) unsigned short sH[16][64];   // 2 KB
    __shared__ alignas(16) float          sXP[16][64];  // 4 KB
    const int lane = threadIdx.x;
    const int nodeBase = blockIdx.x * 16;

    for (int i = lane; i < 64 * 64; i += 32)
        ((unsigned short*)sGW)[i] = f2bf(gat_w[i]);     // gat_w already [c][k] row-major
    for (int i = lane; i < 16 * 64; i += 32) {
        int node = i >> 6, col = i & 63;
        int gn = nodeBase + node; if (gn > N_NODES - 1) gn = N_NODES - 1;
        ((unsigned short*)sH)[i] = f2bf(hbuf[(size_t)gn * HDIM + col]);
    }
    __syncthreads();

    const int halfsel = (lane & 16) ? 8 : 0;
    const int lcol = lane & 15;
    const int mofs = (lane & 16) ? 8 : 0;

    BFrag afrag[2];
#pragma unroll
    for (int c = 0; c < 2; ++c) {
        const int klo = 32 * c + halfsel;
        afrag[c].u[0] = *(const uint4*)&sH[lcol][klo];
        afrag[c].u[1] = *(const uint4*)&sH[lcol][klo + 16];
    }
#pragma unroll
    for (int j = 0; j < 4; ++j) {
        v8f a = {0.f,0.f,0.f,0.f,0.f,0.f,0.f,0.f};
#pragma unroll
        for (int c = 0; c < 2; ++c) {
            BFrag b;
            const int col = 16 * j + lcol;
            const int klo = 32 * c + halfsel;
            b.u[0] = *(const uint4*)&sGW[col][klo];
            b.u[1] = *(const uint4*)&sGW[col][klo + 16];
            a = __builtin_amdgcn_wmma_f32_16x16x32_bf16(
                    false, afrag[c].v, false, b.v, (short)0, a, false, false);
        }
#pragma unroll
        for (int r = 0; r < 8; ++r) {
            const int m = mofs + r, col = 16 * j + lcol;
            sXP[m][col] = a[r];
            int gn = nodeBase + m;
            if (gn < N_NODES) xp[(size_t)gn * HDIM + col] = a[r];
        }
    }
    __syncthreads();
    if (lane < 16) {
        int gn = nodeBase + lane;
        if (gn < N_NODES) {
            float s = 0.f, d = 0.f;
            for (int k = 0; k < HDIM; ++k) {
                float v = sXP[lane][k];
                s += v * att_src[k];
                d += v * att_dst[k];
            }
            a_src[gn] = s;
            a_dst[gn] = d;
        }
    }
}

// ---------------------------------------------------------------------------
// Kernel 3: init accumulators for segment softmax.
// ---------------------------------------------------------------------------
__global__ void init_kernel(float* __restrict__ acc, float* __restrict__ amax,
                            float* __restrict__ denom)
{
    int i = blockIdx.x * blockDim.x + threadIdx.x;
    if (i < N_NODES * HDIM) acc[i] = 0.f;
    if (i < N_NODES) { amax[i] = -3.0e38f; denom[i] = 0.f; }
}

// ---------------------------------------------------------------------------
// Kernel 4: segment max of leaky_relu(a_src[src]+a_dst[dst]) over dst.
// ---------------------------------------------------------------------------
__global__ void alpha_max_kernel(const int* __restrict__ ei,
    const float* __restrict__ a_src, const float* __restrict__ a_dst,
    float* __restrict__ amax)
{
    int e = blockIdx.x * blockDim.x + threadIdx.x;
    if (e >= E_TOT) return;
    int s, d;
    if (e < N_EDGES) { s = ei[e]; d = ei[N_EDGES + e]; }
    else             { s = d = e - N_EDGES; }          // self loops
    float v = a_src[s] + a_dst[d];
    v = v > 0.f ? v : 0.2f * v;
    (void)__hip_atomic_fetch_max(&amax[d], v, __ATOMIC_RELAXED, __HIP_MEMORY_SCOPE_AGENT);
}

// ---------------------------------------------------------------------------
// Kernel 5: ae = exp(alpha - amax[dst]); denom += ae; acc[dst] += ae*xp[src].
// 32 edges per wave; edge metadata broadcast via shuffles, 2 channels/lane.
// ---------------------------------------------------------------------------
__global__ __launch_bounds__(256) void scatter_kernel(const int* __restrict__ ei,
    const float* __restrict__ a_src, const float* __restrict__ a_dst,
    const float* __restrict__ amax, const float* __restrict__ xp,
    float* __restrict__ denom, float* __restrict__ acc)
{
    int e = blockIdx.x * blockDim.x + threadIdx.x;
    int lane = threadIdx.x & 31;
    bool valid = e < E_TOT;
    int s = 0, d = 0; float ae = 0.f;
    if (valid) {
        if (e < N_EDGES) { s = ei[e]; d = ei[N_EDGES + e]; }
        else             { s = d = e - N_EDGES; }
        float v = a_src[s] + a_dst[d];
        v = v > 0.f ? v : 0.2f * v;
        ae = __expf(v - amax[d]);
        atomicAdd(&denom[d], ae);
    }
    unsigned long long mask = __ballot(valid);
    const int c = lane * 2;
    for (int j = 0; j < 32; ++j) {
        if (!((mask >> j) & 1ull)) continue;            // uniform across the wave
        int   sj  = __shfl(s,  j);
        int   dj  = __shfl(d,  j);
        float aej = __shfl(ae, j);
        float2 v = *(const float2*)&xp[(size_t)sj * HDIM + c];
        atomicAdd(&acc[(size_t)dj * HDIM + c    ], aej * v.x);
        atomicAdd(&acc[(size_t)dj * HDIM + c + 1], aej * v.y);
    }
}

// ---------------------------------------------------------------------------
// Kernel 6: normalize, add bias, fc Linear(64,1) + LeakyReLU(0.01).
// ---------------------------------------------------------------------------
__global__ void finalize_kernel(const float* __restrict__ acc,
    const float* __restrict__ denom, const float* __restrict__ gat_bias,
    const float* __restrict__ fc_w, const float* __restrict__ fc_b,
    float* __restrict__ out)
{
    int n = blockIdx.x * blockDim.x + threadIdx.x;
    if (n >= N_NODES) return;
    float inv = __builtin_amdgcn_rcpf(denom[n]);
    float y = fc_b[0];
    for (int c = 0; c < HDIM; ++c)
        y += fc_w[c] * (acc[(size_t)n * HDIM + c] * inv + gat_bias[c]);
    out[n] = y > 0.f ? y : 0.01f * y;
}

extern "C" void kernel_launch(void* const* d_in, const int* in_sizes, int n_in,
                              void* d_out, int out_size, void* d_ws, size_t ws_size,
                              hipStream_t stream)
{
    const float* x        = (const float*)d_in[0];
    const float* W_ih     = (const float*)d_in[1];
    const float* W_hh     = (const float*)d_in[2];
    const float* b_ih     = (const float*)d_in[3];
    const float* b_hh     = (const float*)d_in[4];
    const float* gat_w    = (const float*)d_in[5];
    const float* att_src  = (const float*)d_in[6];
    const float* att_dst  = (const float*)d_in[7];
    const float* gat_bias = (const float*)d_in[8];
    const float* fc_w     = (const float*)d_in[9];
    const float* fc_b     = (const float*)d_in[10];
    const int*   ei       = (const int*)d_in[11];
    float*       out      = (float*)d_out;

    float* ws    = (float*)d_ws;
    float* hacc  = ws;                            // [N,64] h, later reused as acc
    float* xp    = hacc + (size_t)N_NODES * HDIM; // [N,64]
    float* asrc  = xp   + (size_t)N_NODES * HDIM; // [N]
    float* adst  = asrc + N_NODES;                // [N]
    float* amax  = adst + N_NODES;                // [N]
    float* denom = amax + N_NODES;                // [N]

    const int nTiles = (N_NODES + 15) / 16;
    lstm_kernel<<<nTiles, 128, 0, stream>>>(x, W_ih, W_hh, b_ih, b_hh, hacc);
    gat_xp_kernel<<<nTiles, 32, 0, stream>>>(hacc, gat_w, att_src, att_dst, xp, asrc, adst);
    init_kernel<<<(N_NODES * HDIM + 255) / 256, 256, 0, stream>>>(hacc, amax, denom);
    alpha_max_kernel<<<(E_TOT + 255) / 256, 256, 0, stream>>>(ei, asrc, adst, amax);
    scatter_kernel<<<(E_TOT + 255) / 256, 256, 0, stream>>>(ei, asrc, adst, amax, xp, denom, hacc);
    finalize_kernel<<<(N_NODES + 255) / 256, 256, 0, stream>>>(hacc, denom, gat_bias, fc_w, fc_b, out);
}
